// HybridTransformerBlock_8366596292754
// MI455X (gfx1250) — matrix operand order repaired and checked by reading, not compiled, not run on hardware
//
#include <hip/hip_runtime.h>
#include <hip/hip_bf16.h>

typedef _Float16 hf16;
typedef __attribute__((ext_vector_type(8)))  _Float16 v8h;
typedef __attribute__((ext_vector_type(16))) _Float16 v16h;
typedef __attribute__((ext_vector_type(8)))  float    v8f;

#define E_DIM 1024
#define H_NUM 8
#define D_DIM 128
#define T_DIM 2048
#define NTOK  4096   // B*T
#define WIN   256

// ---------------- helpers ----------------
__device__ __forceinline__ v16h ld_afrag(const hf16* __restrict__ p, int lh) {
  // A-fragment (16x32 f16): lane holds row, one contiguous 16-half chunk selected by lane[4]
  return *(const v16h*)(p + lh * 16);
}
__device__ __forceinline__ v8f wmma_f16(v16h a, v16h b, v8f c) {
  return __builtin_amdgcn_wmma_f32_16x16x32_f16(false, a, false, b, (short)0, c, false, false);
}

// ---------------- weight f32 -> f16 ----------------
__global__ void f32_to_f16_kernel(const float* __restrict__ src, hf16* __restrict__ dst, int n) {
  int i = blockIdx.x * blockDim.x + threadIdx.x;
  int stride = gridDim.x * blockDim.x;
  for (; i < n; i += stride) dst[i] = (hf16)src[i];
}

// ---------------- layernorm (fp32 in) -> f16 out ----------------
__global__ void layernorm_to_f16_kernel(const float* __restrict__ x, const float* __restrict__ w,
                                        const float* __restrict__ b, hf16* __restrict__ out) {
  __shared__ float s_sum[256], s_sq[256];
  const int row = blockIdx.x;
  const float* xr = x + (size_t)row * E_DIM;
  float v[4], lsum = 0.f, lsq = 0.f;
#pragma unroll
  for (int i = 0; i < 4; ++i) {
    v[i] = xr[threadIdx.x + i * 256];
    lsum += v[i]; lsq += v[i] * v[i];
  }
  s_sum[threadIdx.x] = lsum; s_sq[threadIdx.x] = lsq;
  __syncthreads();
  for (int off = 128; off > 0; off >>= 1) {
    if (threadIdx.x < off) {
      s_sum[threadIdx.x] += s_sum[threadIdx.x + off];
      s_sq[threadIdx.x]  += s_sq[threadIdx.x + off];
    }
    __syncthreads();
  }
  const float mean = s_sum[0] * (1.0f / E_DIM);
  const float var  = s_sq[0] * (1.0f / E_DIM) - mean * mean;
  const float rstd = rsqrtf(var + 1e-5f);
  hf16* orow = out + (size_t)row * E_DIM;
#pragma unroll
  for (int i = 0; i < 4; ++i) {
    const int c = threadIdx.x + i * 256;
    orow[c] = (hf16)((v[i] - mean) * rstd * w[c] + b[c]);
  }
}

// ---------------- shared GEMM core: wave computes 32(M) x 64(N), K-loop of 32 ----------------
__device__ __forceinline__ void gemm_core(const hf16* __restrict__ A, const hf16* __restrict__ W,
                                          int K, int m_base, int n_base, int ln, int lh,
                                          v8f acc[2][4]) {
  const hf16* ar0 = A + (size_t)(m_base + ln) * K;
  const hf16* ar1 = A + (size_t)(m_base + 16 + ln) * K;
  for (int kk = 0; kk < K; kk += 32) {
    v16h a0 = ld_afrag(ar0 + kk, lh);
    v16h a1 = ld_afrag(ar1 + kk, lh);
#pragma unroll
    for (int t = 0; t < 4; ++t) {
      v16h bf = *(const v16h*)(W + (size_t)(n_base + t * 16 + ln) * K + kk + lh * 16);
      acc[0][t] = wmma_f16(a0, bf, acc[0][t]);
      acc[1][t] = wmma_f16(a1, bf, acc[1][t]);
    }
  }
}

// ---------------- generic WMMA GEMM: C[M,N] = A[M,K] @ W[N,K]^T + bias ----------------
// EPI: 1 = f16 out + ReLU, 2 = f32 out + residual add
template <int EPI>
__global__ void gemm_wmma_kernel(const hf16* __restrict__ A, const hf16* __restrict__ W,
                                 const float* __restrict__ bias, const float* __restrict__ res,
                                 void* __restrict__ out, int M, int N, int K) {
  const int lane = threadIdx.x & 31;
  const int wave = threadIdx.x >> 5;
  const int ln = lane & 15, lh = lane >> 4;
  const int m_base = blockIdx.y * 128 + (wave & 3) * 32;
  const int n_base = blockIdx.x * 128 + (wave >> 2) * 64;

  v8f acc[2][4] = {};
  gemm_core(A, W, K, m_base, n_base, ln, lh, acc);

#pragma unroll
  for (int u = 0; u < 2; ++u) {
#pragma unroll
    for (int t = 0; t < 4; ++t) {
      const int n = n_base + t * 16 + ln;
      const float bv = bias[n];
#pragma unroll
      for (int r = 0; r < 8; ++r) {
        const int m = m_base + u * 16 + r + lh * 8;
        float vv = acc[u][t][r] + bv;
        if (EPI == 1) {
          ((hf16*)out)[(size_t)m * N + n] = (hf16)fmaxf(vv, 0.f);
        } else {
          ((float*)out)[(size_t)m * N + n] = vv + res[(size_t)m * N + n];
        }
      }
    }
  }
}

// ---------------- QKV GEMM: writes q16/k16 token-major, V transposed as vT[B,H,D,T] ----------------
__global__ void gemm_qkv_kernel(const hf16* __restrict__ A, const hf16* __restrict__ W,
                                const float* __restrict__ bias,
                                hf16* __restrict__ q16, hf16* __restrict__ k16,
                                hf16* __restrict__ vT) {
  const int lane = threadIdx.x & 31;
  const int wave = threadIdx.x >> 5;
  const int ln = lane & 15, lh = lane >> 4;
  const int m_base = blockIdx.y * 128 + (wave & 3) * 32;
  const int n_base = blockIdx.x * 128 + (wave >> 2) * 64;
  const int K = E_DIM, N = 3 * E_DIM;

  v8f acc[2][4] = {};
  gemm_core(A, W, K, m_base, n_base, ln, lh, acc);

#pragma unroll
  for (int u = 0; u < 2; ++u) {
#pragma unroll
    for (int t = 0; t < 4; ++t) {
      const int n = n_base + t * 16 + ln;
      const float bv = bias[n];
#pragma unroll
      for (int r = 0; r < 8; ++r) {
        const int m = m_base + u * 16 + r + lh * 8;
        const float vv = acc[u][t][r] + bv;
        if (n < E_DIM) {
          q16[(size_t)m * E_DIM + n] = (hf16)vv;
        } else if (n < 2 * E_DIM) {
          k16[(size_t)m * E_DIM + (n - E_DIM)] = (hf16)vv;
        } else {
          const int d = n - 2 * E_DIM;          // 0..E-1
          const int h = d >> 7, dd = d & 127;   // head, feature-in-head
          const int b = m >> 11, tloc = m & 2047;
          vT[(((size_t)b * H_NUM + h) * D_DIM + dd) * T_DIM + tloc] = (hf16)vv;
        }
      }
    }
  }
}

// ---------------- flash attention, sliding window 256, one wave per 16-query tile ----------------
__global__ void attention_kernel(const hf16* __restrict__ q16, const hf16* __restrict__ k16,
                                 const hf16* __restrict__ vT, hf16* __restrict__ attn_out) {
  __shared__ hf16 ldsP[8 * 16 * 32];  // 1KB per wave P-tile relayout buffer
  const int lane = threadIdx.x & 31;
  const int wave = threadIdx.x >> 5;
  const int ln = lane & 15, lh = lane >> 4;
  const int wid = blockIdx.x * 8 + wave;
  const int qt = wid & 127;
  const int hh = (wid >> 7) & 7;
  const int bb = wid >> 10;
  const int q0 = qt * 16;
  const size_t tokbase = (size_t)bb * T_DIM;

  // preload Q fragments (16x128 = 4 k-steps of 32)
  v16h aq[4];
  {
    const hf16* qrow = q16 + (tokbase + q0 + ln) * E_DIM + hh * D_DIM;
#pragma unroll
    for (int ks = 0; ks < 4; ++ks) aq[ks] = ld_afrag(qrow + ks * 32, lh);
  }
  const hf16* krow_base = k16 + tokbase * E_DIM + hh * D_DIM;
  const hf16* vhead = vT + ((size_t)bb * H_NUM + hh) * D_DIM * T_DIM;

  v8f o[8] = {};
  float mrow[8], lrow[8];
#pragma unroll
  for (int r = 0; r < 8; ++r) { mrow[r] = -1e30f; lrow[r] = 0.f; }

  int ks0 = q0 - WIN; if (ks0 < 0) ks0 = 0; ks0 &= ~31;
  const float scale = 0.08838834764831845f;  // 1/sqrt(128)
  hf16* myP = ldsP + wave * (16 * 32);

  for (int kt = ks0; kt <= q0 + 15; kt += 32) {
    // S = Q @ K^T for 32 keys (two 16-key column tiles); keys always in [0,T)
    v8f s[2] = {};
#pragma unroll
    for (int g = 0; g < 2; ++g) {
      const hf16* krow = krow_base + (size_t)(kt + g * 16 + ln) * E_DIM;
#pragma unroll
      for (int ks = 0; ks < 4; ++ks) {
        v16h bk = *(const v16h*)(krow + ks * 32 + lh * 16);
        s[g] = wmma_f16(aq[ks], bk, s[g]);
      }
    }
    // online softmax (row stats per query row; 16-lane reductions within each half-wave)
#pragma unroll
    for (int r = 0; r < 8; ++r) {
      const int row = q0 + r + lh * 8;
      const int key0 = kt + ln, key1 = kt + 16 + ln;
      const bool a0 = (key0 <= row) && (key0 >= row - WIN);
      const bool a1 = (key1 <= row) && (key1 >= row - WIN);
      float s0 = a0 ? s[0][r] * scale : -1e30f;
      float s1 = a1 ? s[1][r] * scale : -1e30f;
      float rm = fmaxf(s0, s1);
#pragma unroll
      for (int off = 8; off >= 1; off >>= 1) rm = fmaxf(rm, __shfl_xor(rm, off, 32));
      const float mnew  = fmaxf(mrow[r], rm);
      const float alpha = __expf(mrow[r] - mnew);
      mrow[r] = mnew;
      float p0 = a0 ? __expf(s0 - mnew) : 0.f;
      float p1 = a1 ? __expf(s1 - mnew) : 0.f;
      float psum = p0 + p1;
#pragma unroll
      for (int off = 8; off >= 1; off >>= 1) psum += __shfl_xor(psum, off, 32);
      lrow[r] = lrow[r] * alpha + psum;
#pragma unroll
      for (int t = 0; t < 8; ++t) o[t][r] *= alpha;
      hf16* prow = myP + (r + lh * 8) * 32;  // C-layout -> row-major LDS tile
      prow[ln]      = (hf16)p0;
      prow[16 + ln] = (hf16)p1;
    }
    asm volatile("s_wait_dscnt 0" ::: "memory");
    // reload P in A-fragment layout
    v16h pa = ld_afrag(myP + ln * 32, lh);
    // O += P @ V : B fragments are contiguous loads from transposed V
#pragma unroll
    for (int t = 0; t < 8; ++t) {
      v16h bv = *(const v16h*)(vhead + (size_t)(t * 16 + ln) * T_DIM + kt + lh * 16);
      o[t] = wmma_f16(pa, bv, o[t]);
    }
  }
  // normalize and store (token-major [B,T,H*D])
#pragma unroll
  for (int r = 0; r < 8; ++r) {
    const float inv = 1.0f / lrow[r];
    const size_t orow = (tokbase + q0 + r + lh * 8) * E_DIM + hh * D_DIM;
#pragma unroll
    for (int t = 0; t < 8; ++t)
      attn_out[orow + t * 16 + ln] = (hf16)(o[t][r] * inv);
  }
}

extern "C" void kernel_launch(void* const* d_in, const int* in_sizes, int n_in,
                              void* d_out, int out_size, void* d_ws, size_t ws_size,
                              hipStream_t stream) {
  (void)in_sizes; (void)n_in; (void)out_size; (void)ws_size;
  const float* x     = (const float*)d_in[0];
  const float* ln1_w = (const float*)d_in[1];
  const float* ln1_b = (const float*)d_in[2];
  const float* ln2_w = (const float*)d_in[3];
  const float* ln2_b = (const float*)d_in[4];
  const float* in_w  = (const float*)d_in[5];
  const float* in_b  = (const float*)d_in[6];
  const float* out_w = (const float*)d_in[7];
  const float* out_b = (const float*)d_in[8];
  const float* w1    = (const float*)d_in[9];
  const float* b1    = (const float*)d_in[10];
  const float* w2    = (const float*)d_in[11];
  const float* b2    = (const float*)d_in[12];

  char* ws = (char*)d_ws;
  size_t off = 0;
  auto alloc = [&](size_t bytes) {
    char* p = ws + off;
    off += (bytes + 255) & ~(size_t)255;
    return (void*)p;
  };
  hf16*  h16     = (hf16*)alloc((size_t)NTOK * E_DIM * 2);
  hf16*  w_in16  = (hf16*)alloc((size_t)3 * E_DIM * E_DIM * 2);
  hf16*  w_out16 = (hf16*)alloc((size_t)E_DIM * E_DIM * 2);
  hf16*  w1_16   = (hf16*)alloc((size_t)4 * E_DIM * E_DIM * 2);
  hf16*  w2_16   = (hf16*)alloc((size_t)4 * E_DIM * E_DIM * 2);
  hf16*  q16     = (hf16*)alloc((size_t)NTOK * E_DIM * 2);
  hf16*  k16     = (hf16*)alloc((size_t)NTOK * E_DIM * 2);
  hf16*  vT      = (hf16*)alloc((size_t)NTOK * E_DIM * 2);
  hf16*  attn16  = (hf16*)alloc((size_t)NTOK * E_DIM * 2);
  float* h2      = (float*)alloc((size_t)NTOK * E_DIM * 4);
  hf16*  m16     = (hf16*)alloc((size_t)NTOK * E_DIM * 2);
  hf16*  a1      = (hf16*)alloc((size_t)NTOK * 4 * E_DIM * 2);

  f32_to_f16_kernel<<<1024, 256, 0, stream>>>(in_w,  w_in16,  3 * E_DIM * E_DIM);
  f32_to_f16_kernel<<<1024, 256, 0, stream>>>(out_w, w_out16, E_DIM * E_DIM);
  f32_to_f16_kernel<<<1024, 256, 0, stream>>>(w1,    w1_16,   4 * E_DIM * E_DIM);
  f32_to_f16_kernel<<<1024, 256, 0, stream>>>(w2,    w2_16,   4 * E_DIM * E_DIM);

  // LN1 -> h16
  layernorm_to_f16_kernel<<<NTOK, 256, 0, stream>>>(x, ln1_w, ln1_b, h16);
  // qkv projection; V written transposed
  gemm_qkv_kernel<<<dim3(3 * E_DIM / 128, NTOK / 128), 256, 0, stream>>>(
      h16, w_in16, in_b, q16, k16, vT);
  // sliding-window attention
  attention_kernel<<<256, 256, 0, stream>>>(q16, k16, vT, attn16);
  // h2 = x + attn @ out_w^T + out_b
  gemm_wmma_kernel<2><<<dim3(E_DIM / 128, NTOK / 128), 256, 0, stream>>>(
      attn16, w_out16, out_b, x, h2, NTOK, E_DIM, E_DIM);
  // LN2 -> m16
  layernorm_to_f16_kernel<<<NTOK, 256, 0, stream>>>(h2, ln2_w, ln2_b, m16);
  // a1 = relu(m @ w1^T + b1)
  gemm_wmma_kernel<1><<<dim3(4 * E_DIM / 128, NTOK / 128), 256, 0, stream>>>(
      m16, w1_16, b1, nullptr, a1, NTOK, 4 * E_DIM, E_DIM);
  // out = h2 + a1 @ w2^T + b2
  gemm_wmma_kernel<2><<<dim3(E_DIM / 128, NTOK / 128), 256, 0, stream>>>(
      a1, w2_16, b2, h2, d_out, NTOK, E_DIM, 4 * E_DIM);
}
